// LSTMModel_5634997092553
// MI455X (gfx1250) — matrix operand order repaired and checked by reading, not compiled, not run on hardware
//
#include <hip/hip_runtime.h>

typedef __attribute__((ext_vector_type(16))) _Float16 v16h;
typedef __attribute__((ext_vector_type(8)))  _Float16 v8h;
typedef __attribute__((ext_vector_type(8)))  float    v8f;
typedef __attribute__((ext_vector_type(2)))  float    v2f;

#define TT    512
#define HH    32
#define BATCH 8192
#define WPB   8      // waves per block
#define CHUNK 64     // timesteps of x staged in LDS per refill

__device__ __forceinline__ float fast_sig(float x) {
    // 1 / (1 + 2^(-x*log2(e)))
    float e = __builtin_amdgcn_exp2f(-1.44269504f * x);
    return __builtin_amdgcn_rcpf(1.0f + e);
}
__device__ __forceinline__ float fast_tanh(float x) {
    // 2 / (1 + 2^(-2x*log2(e))) - 1
    float e = __builtin_amdgcn_exp2f(-2.88539008f * x);
    return 2.0f * __builtin_amdgcn_rcpf(1.0f + e) - 1.0f;
}

__global__ __launch_bounds__(256)
void lstm_wmma_kernel(const float* __restrict__ x,
                      const float* __restrict__ W_ih,
                      const float* __restrict__ W_hh,
                      const float* __restrict__ b_ih,
                      const float* __restrict__ b_hh,
                      const float* __restrict__ W_fc,
                      const float* __restrict__ b_fc,
                      float* __restrict__ out)
{
    // per-wave LDS slices
    __shared__ float    xbuf[WPB][CHUNK * 16];   // x transposed: [t][m], pitch 16 floats
    __shared__ _Float16 hbuf[WPB][16 * 40];      // h: [m][j], pitch 40 halves (80B: 16B-aligned, bank-spread)

    const int tid   = threadIdx.x;
    const int w     = tid >> 5;          // wave slot in block
    const int lane  = tid & 31;
    const int n     = lane & 15;         // column within 16-wide tile
    const int hi    = lane >> 4;         // half-wave select
    const int wbase = (blockIdx.x * WPB + w) * 16;   // batch row base for this wave

    // ---- B fragments: W_hh (gate-major rows) -> f16, B layout, contiguous k (k = 16*hi + e) ----
    // B[k][g] = W_hh[g][k]; lane holds col g = 16*tile + n, K = hi*16 + e (e=0..15)
    v16h bfrag[8];
#pragma unroll
    for (int tIx = 0; tIx < 8; ++tIx) {
        const float* wr = W_hh + (16 * tIx + n) * HH + hi * 16;
#pragma unroll
        for (int e = 0; e < 16; ++e) bfrag[tIx][e] = (_Float16)wr[e];
    }

    // per-lane input-projection constants (explicit broadcast pairs -> exactly 32 VGPRs)
    v2f wih2[8], bs2[8];
#pragma unroll
    for (int tIx = 0; tIx < 8; ++tIx) {
        int g = 16 * tIx + n;
        float wv = W_ih[g];
        float bv = b_ih[g] + b_hh[g];
        wih2[tIx] = (v2f){wv, wv};
        bs2 [tIx] = (v2f){bv, bv};
    }

    v16h afrag = {};            // h in A layout; zero for t=0
    float cst[16];              // c state, D layout: [p*8 + r]
#pragma unroll
    for (int i = 0; i < 16; ++i) cst[i] = 0.0f;

#pragma clang loop unroll(disable)
    for (int t = 0; t < TT; ++t) {
        if ((t & (CHUNK - 1)) == 0) {
            // stage x[wbase + 0..15][t .. t+63] into LDS transposed [tt][m]
            const float* xg = x + (size_t)(wbase + n) * TT + t + hi * 32;
#pragma unroll
            for (int k = 0; k < 8; ++k) {
                float4 v = ((const float4*)xg)[k];
                int ts = hi * 32 + k * 4;
                xbuf[w][(ts + 0) * 16 + n] = v.x;
                xbuf[w][(ts + 1) * 16 + n] = v.y;
                xbuf[w][(ts + 2) * 16 + n] = v.z;
                xbuf[w][(ts + 3) * 16 + n] = v.w;
            }
            if (t + CHUNK < TT)               // pull next chunk toward L2/L0 early
                __builtin_prefetch(xg + CHUNK, 0, 0);
            __builtin_amdgcn_wave_barrier();
        }
        const int tt = t & (CHUNK - 1);

        // per-lane x values for its 8 batch rows (m = hi*8 + r); broadcast LDS read
        v8f xv = *(const v8f*)&xbuf[w][tt * 16 + hi * 8];
        v2f xp0 = (v2f){xv[0], xv[1]};
        v2f xp1 = (v2f){xv[2], xv[3]};
        v2f xp2 = (v2f){xv[4], xv[5]};
        v2f xp3 = (v2f){xv[6], xv[7]};

        // two half-steps: p selects column group j = 16*p + n; tiles {p, 2+p, 4+p, 6+p}
#pragma unroll
        for (int p = 0; p < 2; ++p) {
            v8f acc[4];                       // only 4 accumulator tiles live at once
#pragma unroll
            for (int gi = 0; gi < 4; ++gi) {  // gate order i,f,g,o
                const int tIx = gi * 2 + p;
                union { v8f v; v2f p2[4]; } cc;
                cc.p2[0] = xp0 * wih2[tIx] + bs2[tIx];
                cc.p2[1] = xp1 * wih2[tIx] + bs2[tIx];
                cc.p2[2] = xp2 * wih2[tIx] + bs2[tIx];
                cc.p2[3] = xp3 * wih2[tIx] + bs2[tIx];
                acc[gi] = __builtin_amdgcn_wmma_f32_16x16x32_f16(
                    false, afrag, false, bfrag[tIx], (short)0, cc.v, false, false);
            }
            // activations + state update; h -> LDS as f16 (D layout scatter)
#pragma unroll
            for (int r = 0; r < 8; ++r) {
                float ig = fast_sig (acc[0][r]);
                float fg = fast_sig (acc[1][r]);
                float gg = fast_tanh(acc[2][r]);
                float og = fast_sig (acc[3][r]);
                float cv = fg * cst[p * 8 + r] + ig * gg;
                cst[p * 8 + r] = cv;
                float hv = og * fast_tanh(cv);
                hbuf[w][(hi * 8 + r) * 40 + 16 * p + n] = (_Float16)hv;
            }
        }
        __builtin_amdgcn_wave_barrier();

        // reload h in A layout: row m = n; K = hi*16+{0..15}
        {
            v8h* ap = (v8h*)&afrag;
            ap[0] = *(const v8h*)&hbuf[w][n * 40 + hi * 16];
            ap[1] = *(const v8h*)&hbuf[w][n * 40 + hi * 16 + 8];
        }
    }

    // ---- head: out[m] = sum_j h[m][j] * W_fc[j] + b_fc ----
    float wfc0 = W_fc[n], wfc1 = W_fc[16 + n];
#pragma unroll
    for (int r = 0; r < 8; ++r) {
        int m = hi * 8 + r;
        float h0 = (float)hbuf[w][m * 40 + n];
        float h1 = (float)hbuf[w][m * 40 + 16 + n];
        xbuf[w][m * 16 + n] = h0 * wfc0 + h1 * wfc1;   // reuse xbuf as f32 scratch
    }
    __builtin_amdgcn_wave_barrier();
    if (lane < 16) {
        float s = 0.0f;
#pragma unroll
        for (int k = 0; k < 16; ++k) s += xbuf[w][lane * 16 + k];
        out[wbase + lane] = s + b_fc[0];
    }
}

extern "C" void kernel_launch(void* const* d_in, const int* in_sizes, int n_in,
                              void* d_out, int out_size, void* d_ws, size_t ws_size,
                              hipStream_t stream) {
    const float* x    = (const float*)d_in[0];
    const float* W_ih = (const float*)d_in[1];
    const float* W_hh = (const float*)d_in[2];
    const float* b_ih = (const float*)d_in[3];
    const float* b_hh = (const float*)d_in[4];
    const float* W_fc = (const float*)d_in[5];
    const float* b_fc = (const float*)d_in[6];
    float* out = (float*)d_out;

    dim3 grid(BATCH / (WPB * 16));   // 64 blocks
    dim3 block(WPB * 32);            // 256 threads = 8 waves
    lstm_wmma_kernel<<<grid, block, 0, stream>>>(x, W_ih, W_hh, b_ih, b_hh, W_fc, b_fc, out);
}
